// BMMGroupedGEMM_27874337751306
// MI455X (gfx1250) — compile-verified, matrix operand order checked by reading, not run on hardware
//
#include <hip/hip_runtime.h>
#include <stdint.h>

typedef __attribute__((ext_vector_type(16))) __bf16    v16bf;
typedef __attribute__((ext_vector_type(8)))  float     v8f;
typedef __attribute__((ext_vector_type(4)))  unsigned  v4u;
typedef __attribute__((ext_vector_type(4)))  int       v4i_t;
typedef unsigned short ushort_t;

typedef __attribute__((address_space(1))) v4i_t gv4i_t;   // global int4
typedef __attribute__((address_space(3))) v4i_t lv4i_t;   // LDS int4

#define NE     8
#define MAXM   2048
#define KDIM   2048
#define NDIM   8192
#define BM     128
#define BN     128
#define BK     64
#define NKT    (KDIM / BK)   // 32 K-steps
#define LDSP   72            // padded LDS row stride (bf16 elems) = 144 bytes

#if __has_builtin(__builtin_amdgcn_global_load_async_to_lds_b128)
#define HAS_ASYNC_LDS 1
#else
#define HAS_ASYNC_LDS 0
#endif

#if __has_builtin(__builtin_amdgcn_s_wait_asynccnt)
#define WAIT_ASYNC() __builtin_amdgcn_s_wait_asynccnt(0)
#else
#define WAIT_ASYNC() asm volatile("s_wait_asynccnt 0x0" ::: "memory")
#endif

union FragU { v4u u[2]; v16bf b; };

// round-to-nearest-even f32 -> bf16 pair packed into one dword (pre-pass only)
__device__ __forceinline__ unsigned pack2_bf16(float lo, float hi) {
  unsigned ulo = __builtin_bit_cast(unsigned, lo);
  unsigned uhi = __builtin_bit_cast(unsigned, hi);
  ulo = (ulo + 0x7FFFu + ((ulo >> 16) & 1u)) >> 16;
  uhi = (uhi + 0x7FFFu + ((uhi >> 16) & 1u)) & 0xFFFF0000u;
  return ulo | uhi;
}

__device__ __forceinline__ uint4 cvt8(const float4 f0, const float4 f1) {
  uint4 o;
  o.x = pack2_bf16(f0.x, f0.y);
  o.y = pack2_bf16(f0.z, f0.w);
  o.z = pack2_bf16(f1.x, f1.y);
  o.w = pack2_bf16(f1.z, f1.w);
  return o;
}

// ---- pre-pass: x f32 -> bf16, rows m >= m_sizes[e] zeroed (mask folded in here) ----
__global__ __launch_bounds__(256)
void convert_x_bf16(const float* __restrict__ X, const int* __restrict__ MS,
                    ushort_t* __restrict__ XB) {
  const size_t i = ((size_t)blockIdx.x * 256 + threadIdx.x) * 8;
  const int e = (int)(i / ((size_t)MAXM * KDIM));
  const int m = (int)((i / KDIM) % MAXM);
  float4 f0 = *(const float4*)(X + i);
  float4 f1 = *(const float4*)(X + i + 4);
  if (m >= MS[e]) {
    f0 = make_float4(0.f, 0.f, 0.f, 0.f);
    f1 = f0;
  }
  *(uint4*)(XB + i) = cvt8(f0, f1);
}

// ---- pre-pass: W f32 -> bf16 ----
__global__ __launch_bounds__(256)
void convert_w_bf16(const float* __restrict__ W, ushort_t* __restrict__ WB) {
  const size_t i = ((size_t)blockIdx.x * 256 + threadIdx.x) * 8;
  float4 f0 = *(const float4*)(W + i);
  float4 f1 = *(const float4*)(W + i + 4);
  *(uint4*)(WB + i) = cvt8(f0, f1);
}

// ---- main grouped GEMM: bf16 tiles via async global->LDS, f32 WMMA accumulation ----
__global__ __launch_bounds__(256)
void grouped_gemm_bf16_wmma(const ushort_t* __restrict__ XB,
                            const ushort_t* __restrict__ WB,
                            float* __restrict__ Out) {
  __shared__ __align__(16) ushort_t As[2][BM][LDSP];
  __shared__ __align__(16) ushort_t Bs[2][BM][LDSP];

  const int bid = blockIdx.x;
  const int mt  = bid & 15;          // M fastest: weight N-slab stays hot in L2
  const int nt  = (bid >> 4) & 63;
  const int e   = bid >> 10;

  const int m0 = mt * BM;
  const int n0 = nt * BN;

  const ushort_t* Xe = XB + (size_t)e * MAXM * KDIM + (size_t)m0 * KDIM;
  const ushort_t* We = WB + (size_t)e * NDIM * KDIM + (size_t)n0 * KDIM;
  float*          Oe = Out + (size_t)e * MAXM * NDIM + (size_t)m0 * NDIM + n0;

  const int tid  = threadIdx.x;
  const int lane = tid & 31;
  const int wave = tid >> 5;
  const int wm   = (wave & 3) * 32;   // 4 waves over M (32 rows each)
  const int wn   = (wave >> 2) * 64;  // 2 waves over N (64 cols each)

  const int lrow = lane & 15;
  const int lhi  = lane >> 4;
  const int koff = lhi * 8;

  // copy mapping: tile = 128 rows x 64 bf16 = 1024 x 16B chunks; 256 threads x 4 chunks
  int crow[4], ccol[4], goff[4];
#pragma unroll
  for (int j = 0; j < 4; ++j) {
    const int idx = tid + j * 256;
    crow[j] = idx >> 3;
    ccol[j] = (idx & 7) * 8;                 // bf16 element column inside BK
    goff[j] = crow[j] * KDIM + ccol[j];      // element offset, k-invariant part
  }

  // issue global->LDS copies of the (kt) K-slab into LDS stage `buf`
  auto copy_tile = [&](int kt, int buf) {
    const int kb = kt * BK;
#pragma unroll
    for (int j = 0; j < 4; ++j) {
      const ushort_t* ga = Xe + goff[j] + kb;
      const ushort_t* gb = We + goff[j] + kb;
      ushort_t* la = &As[buf][crow[j]][ccol[j]];
      ushort_t* lb = &Bs[buf][crow[j]][ccol[j]];
#if HAS_ASYNC_LDS
      __builtin_amdgcn_global_load_async_to_lds_b128((gv4i_t*)ga, (lv4i_t*)la, 0, 0);
      __builtin_amdgcn_global_load_async_to_lds_b128((gv4i_t*)gb, (lv4i_t*)lb, 0, 0);
#else
      *(uint4*)la = *(const uint4*)ga;       // sync fallback: no staging arrays -> no spills
      *(uint4*)lb = *(const uint4*)gb;
#endif
    }
  };

  // ISA 16-bit A/B fragment: lane<16 holds K{kb+0..7, kb+16..23}, lane>=16 K{kb+8..15, kb+24..31}
  auto load_frag = [&](const ushort_t (*T)[LDSP], int rowbase, int kb) -> v16bf {
    FragU f;
    const ushort_t* p = &T[rowbase + lrow][kb + koff];
    f.u[0] = *(const v4u*)(p);
    f.u[1] = *(const v4u*)(p + 16);
    return f.b;
  };

  v8f acc[2][4] = {};

  copy_tile(0, 0);
  WAIT_ASYNC();
  __syncthreads();

  for (int kt = 0; kt < NKT; ++kt) {
    const int buf = kt & 1;
    if (kt + 1 < NKT) copy_tile(kt + 1, buf ^ 1);

#pragma unroll
    for (int kc = 0; kc < 2; ++kc) {
      const int kb = kc * 32;
      v16bf a0 = load_frag(As[buf], wm,      kb);
      v16bf a1 = load_frag(As[buf], wm + 16, kb);
      v16bf b0 = load_frag(Bs[buf], wn,      kb);
      v16bf b1 = load_frag(Bs[buf], wn + 16, kb);
      v16bf b2 = load_frag(Bs[buf], wn + 32, kb);
      v16bf b3 = load_frag(Bs[buf], wn + 48, kb);

      acc[0][0] = __builtin_amdgcn_wmma_f32_16x16x32_bf16(false, a0, false, b0, (short)0, acc[0][0], false, false);
      acc[0][1] = __builtin_amdgcn_wmma_f32_16x16x32_bf16(false, a0, false, b1, (short)0, acc[0][1], false, false);
      acc[0][2] = __builtin_amdgcn_wmma_f32_16x16x32_bf16(false, a0, false, b2, (short)0, acc[0][2], false, false);
      acc[0][3] = __builtin_amdgcn_wmma_f32_16x16x32_bf16(false, a0, false, b3, (short)0, acc[0][3], false, false);
      acc[1][0] = __builtin_amdgcn_wmma_f32_16x16x32_bf16(false, a1, false, b0, (short)0, acc[1][0], false, false);
      acc[1][1] = __builtin_amdgcn_wmma_f32_16x16x32_bf16(false, a1, false, b1, (short)0, acc[1][1], false, false);
      acc[1][2] = __builtin_amdgcn_wmma_f32_16x16x32_bf16(false, a1, false, b2, (short)0, acc[1][2], false, false);
      acc[1][3] = __builtin_amdgcn_wmma_f32_16x16x32_bf16(false, a1, false, b3, (short)0, acc[1][3], false, false);
    }

    if (kt + 1 < NKT) {
      WAIT_ASYNC();
      __syncthreads();
    }
  }

  // epilogue: C/D layout — VGPR r: lanes 0-15 -> M=r, lanes 16-31 -> M=r+8; N = lane%16
#pragma unroll
  for (int am = 0; am < 2; ++am) {
#pragma unroll
    for (int bn = 0; bn < 4; ++bn) {
      const int rbase = wm + am * 16 + lhi * 8;
      const int col   = wn + bn * 16 + lrow;
#pragma unroll
      for (int r = 0; r < 8; ++r) {
        Oe[(size_t)(rbase + r) * NDIM + col] = acc[am][bn][r];
      }
    }
  }
}

extern "C" void kernel_launch(void* const* d_in, const int* in_sizes, int n_in,
                              void* d_out, int out_size, void* d_ws, size_t ws_size,
                              hipStream_t stream) {
  const float* x   = (const float*)d_in[0];  // [E, MAXM, K] f32
  const float* w   = (const float*)d_in[1];  // [E, N, K]    f32
  const int*   ms  = (const int*)d_in[2];    // [E]
  float*       out = (float*)d_out;          // [E, MAXM, N] f32

  ushort_t* xb = (ushort_t*)d_ws;                 // [E, MAXM, K] bf16
  ushort_t* wb = xb + (size_t)NE * MAXM * KDIM;   // [E, N, K]    bf16

  const int xblocks = (int)(((size_t)NE * MAXM * KDIM) / (256 * 8));  // 16384
  const int wblocks = (int)(((size_t)NE * NDIM * KDIM) / (256 * 8));  // 65536
  convert_x_bf16<<<dim3(xblocks), dim3(256), 0, stream>>>(x, ms, xb);
  convert_w_bf16<<<dim3(wblocks), dim3(256), 0, stream>>>(w, wb);

  const int nblocks = NE * (MAXM / BM) * (NDIM / BN);  // 8192
  grouped_gemm_bf16_wmma<<<dim3(nblocks), dim3(256), 0, stream>>>(xb, wb, out);
}
